// vgg11_37580963840725
// MI455X (gfx1250) — 8-bit matrix operand order repaired and checked by reading, compiled, not run on hardware
//
#include <hip/hip_runtime.h>
#include <stdint.h>


typedef __attribute__((ext_vector_type(16))) int   v16i;
typedef __attribute__((ext_vector_type(8)))  float v8f;

#define EPSBN  1e-3f
#define FP8_P1 ((uint8_t)0x38)
#define FP8_M1 ((uint8_t)0xB8)

__global__ void stage1_kernel(const float* __restrict__ in,
                              const float* __restrict__ w,
                              const float* __restrict__ bias,
                              const float* __restrict__ bn,
                              uint8_t* __restrict__ out)
{
    int c = threadIdx.x;
    int p = blockIdx.x;
    int b = p / (112 * 112);
    int r = p % (112 * 112);
    int py = r / 112, px = r % 112;

    float g  = bn[c], be = bn[64 + c], mu = bn[128 + c], va = bn[192 + c];
    float iv = rsqrtf(va + EPSBN);
    float bi = bias[c];

    float mx = -1e30f;
    #pragma unroll
    for (int dy = 0; dy < 2; ++dy)
    #pragma unroll
    for (int dx = 0; dx < 2; ++dx) {
        int oy = 2 * py + dy, ox = 2 * px + dx;
        float acc = bi;
        #pragma unroll
        for (int ky = 0; ky < 3; ++ky) {
            int iy = oy + ky - 1;
            if (iy < 0 || iy >= 224) continue;
            #pragma unroll
            for (int kx = 0; kx < 3; ++kx) {
                int ix = ox + kx - 1;
                if (ix < 0 || ix >= 224) continue;
                const float* ip = in + ((size_t)(b * 224 + iy) * 224 + ix) * 3;
                const float* wp = w + (ky * 3 + kx) * 192 + c;
                acc += ip[0] * wp[0] + ip[1] * wp[64] + ip[2] * wp[128];
            }
        }
        acc = fmaxf(acc, 0.f);
        float v = g * (acc - mu) * iv + be;
        mx = fmaxf(mx, v);
    }
    out[(size_t)p * 64 + c] = (mx >= 0.f) ? FP8_P1 : FP8_M1;
}

__global__ void packw_kernel(const float* __restrict__ w, uint8_t* __restrict__ wp,
                             int cin, int cinLog2, int cout, int kpad)
{
    int k = blockIdx.x * 256 + threadIdx.x;
    int n = blockIdx.y;
    if (k >= kpad) return;
    uint8_t byte = 0;
    if (k < 9 * cin) {
        int t  = k >> cinLog2;
        int ci = k & (cin - 1);
        byte = (w[(size_t)(t * cin + ci) * cout + n] >= 0.f) ? FP8_P1 : FP8_M1;
    }
    wp[(size_t)n * kpad + k] = byte;
}

template <bool FP32OUT>
__global__ __launch_bounds__(256)
void bconv_kernel(const uint8_t* __restrict__ in, const uint8_t* __restrict__ wp,
                  const float* __restrict__ bn,
                  uint8_t* __restrict__ out8, float* __restrict__ outf,
                  int H, int W, int cinLog2, int Cout, int Kpad, int Mtot)
{
    const int Cin = 1 << cinLog2;
    const int HW  = H * W;

    __shared__ __align__(16) uint8_t As[2][128 * 128];
    __shared__ __align__(16) uint8_t Bs[2][64 * 128];

    const int tid   = threadIdx.x;
    const int tileM = blockIdx.x;
    const int tileN = blockIdx.y;

    const int ksg = (tid & 7) * 16;

    int aPl[4], aB[4], aY[4], aX[4];
    bool aOK[4];
    #pragma unroll
    for (int i = 0; i < 4; ++i) {
        int pl = (tid >> 3) + 32 * i;
        int p  = tileM * 128 + pl;
        aPl[i] = pl;
        aOK[i] = (p < Mtot);
        int pp = aOK[i] ? p : 0;
        aB[i] = pp / HW;
        int r = pp % HW;
        aY[i] = r / W;
        aX[i] = r % W;
    }
    const int brow = tid >> 3;
    const uint8_t* wrow0 = wp + (size_t)(tileN * 64 + brow) * Kpad;
    const uint8_t* wrow1 = wrow0 + (size_t)32 * Kpad;

    const int lane = tid & 31;
    const int wave = tid >> 5;
    const int mP   = (wave & 3) * 2;
    const int nP   = (wave >> 2) * 2;
    const int hf   = lane >> 4;
    const int l15  = lane & 15;

    const v8f vz = {0.f, 0.f, 0.f, 0.f, 0.f, 0.f, 0.f, 0.f};
    v8f acc00 = vz, acc01 = vz, acc10 = vz, acc11 = vz;

    auto stage = [&](int ch, int buf) {
        int k   = ch * 128 + ksg;
        int tap = k >> cinLog2;
        int ci  = k & (Cin - 1);
        int ky  = tap / 3, kx = tap % 3;
        #pragma unroll
        for (int i = 0; i < 4; ++i) {
            uint4 d = {0u, 0u, 0u, 0u};
            if (tap < 9 && aOK[i]) {
                int iy = aY[i] + ky - 1, ix = aX[i] + kx - 1;
                if (iy >= 0 && iy < H && ix >= 0 && ix < W)
                    d = *(const uint4*)(in + (((size_t)(aB[i] * H + iy) * W + ix)
                                              << cinLog2) + ci);
            }
            *(uint4*)(As[buf] + aPl[i] * 128 + ksg) = d;
        }
        *(uint4*)(Bs[buf] + brow * 128 + ksg)        = *(const uint4*)(wrow0 + ch * 128 + ksg);
        *(uint4*)(Bs[buf] + (brow + 32) * 128 + ksg) = *(const uint4*)(wrow1 + ch * 128 + ksg);
    };

    const int nCh = Kpad >> 7;
    stage(0, 0);
    for (int ch = 0; ch < nCh; ++ch) {
        __syncthreads();
        if (ch + 1 < nCh) stage(ch + 1, (ch + 1) & 1);

        const uint8_t* Ab = As[ch & 1];
        const uint8_t* Bb = Bs[ch & 1];

        v16i a0, a1;
        {
            const uint8_t* p0 = Ab + (mP * 16 + l15) * 128 + 16 * hf;
            const uint8_t* p1 = p0 + 16 * 128;
            #pragma unroll
            for (int j = 0; j < 8; ++j) {
                uint2 q0 = *(const uint2*)(p0 + 32 * (j >> 1) + 8 * (j & 1));
                uint2 q1 = *(const uint2*)(p1 + 32 * (j >> 1) + 8 * (j & 1));
                a0[2 * j] = (int)q0.x; a0[2 * j + 1] = (int)q0.y;
                a1[2 * j] = (int)q1.x; a1[2 * j + 1] = (int)q1.y;
            }
        }
        v16i b0, b1;
        {
            const uint8_t* p0 = Bb + (nP * 16 + l15) * 128 + 16 * hf;
            const uint8_t* p1 = p0 + 16 * 128;
            #pragma unroll
            for (int g = 0; g < 4; ++g) {
                uint4 q0 = *(const uint4*)(p0 + 32 * g);
                uint4 q1 = *(const uint4*)(p1 + 32 * g);
                b0[4 * g] = (int)q0.x; b0[4 * g + 1] = (int)q0.y;
                b0[4 * g + 2] = (int)q0.z; b0[4 * g + 3] = (int)q0.w;
                b1[4 * g] = (int)q1.x; b1[4 * g + 1] = (int)q1.y;
                b1[4 * g + 2] = (int)q1.z; b1[4 * g + 3] = (int)q1.w;
            }
        }
        acc00 = __builtin_amdgcn_wmma_f32_16x16x128_fp8_fp8(a0, b0, (short)0, acc00, false, false);
        acc01 = __builtin_amdgcn_wmma_f32_16x16x128_fp8_fp8(a0, b1, (short)0, acc01, false, false);
        acc10 = __builtin_amdgcn_wmma_f32_16x16x128_fp8_fp8(a1, b0, (short)0, acc10, false, false);
        acc11 = __builtin_amdgcn_wmma_f32_16x16x128_fp8_fp8(a1, b1, (short)0, acc11, false, false);
    }

    #pragma unroll
    for (int j = 0; j < 2; ++j) {
        const int c = tileN * 64 + (nP + j) * 16 + l15;
        float g  = bn[c], be = bn[Cout + c], mu = bn[2 * Cout + c], va = bn[3 * Cout + c];
        float iv = rsqrtf(va + EPSBN);
        #pragma unroll
        for (int i = 0; i < 2; ++i) {
            const v8f& a = (i == 0) ? (j == 0 ? acc00 : acc01)
                                    : (j == 0 ? acc10 : acc11);
            #pragma unroll
            for (int r = 0; r < 8; ++r) {
                int p = tileM * 128 + (mP + i) * 16 + 8 * hf + r;
                if (p < Mtot) {
                    float v = fmaxf(a[r], 0.f);
                    v = g * (v - mu) * iv + be;
                    if (FP32OUT) {
                        outf[(size_t)p * Cout + c] = v;
                    } else {
                        out8[(size_t)p * Cout + c] = (v >= 0.f) ? FP8_P1 : FP8_M1;
                    }
                }
            }
        }
    }
}

__global__ void pool8_kernel(const uint8_t* __restrict__ in, uint8_t* __restrict__ out,
                             int Ho, int Wo, int C, size_t total)
{
    size_t idx = (size_t)blockIdx.x * 256 + threadIdx.x;
    if (idx >= total) return;
    int    c = (int)(idx % C);
    size_t p = idx / C;
    int    x = (int)(p % Wo); p /= Wo;
    int    y = (int)(p % Ho); p /= Ho;
    int    b = (int)p;
    int Wi = Wo * 2;
    const uint8_t* base = in + ((size_t)((b * Ho * 2 + 2 * y) * Wi + 2 * x)) * C + c;
    uint8_t a0 = base[0];
    uint8_t a1 = base[C];
    uint8_t a2 = base[(size_t)Wi * C];
    uint8_t a3 = base[(size_t)Wi * C + C];
    out[idx] = (a0 == FP8_P1 || a1 == FP8_P1 || a2 == FP8_P1 || a3 == FP8_P1)
                   ? FP8_P1 : FP8_M1;
}

__global__ void gap_kernel(const float* __restrict__ y, uint8_t* __restrict__ vec)
{
    int idx = blockIdx.x * 256 + threadIdx.x;
    if (idx >= 16 * 512) return;
    int c = idx & 511;
    int b = idx >> 9;
    float s = 0.f;
    for (int py = 0; py < 7; ++py)
        for (int px = 0; px < 7; ++px) {
            float m = -1e30f;
            #pragma unroll
            for (int dy = 0; dy < 2; ++dy)
            #pragma unroll
            for (int dx = 0; dx < 2; ++dx)
                m = fmaxf(m, y[((size_t)((b * 14 + 2 * py + dy) * 14) +
                                2 * px + dx) * 512 + c]);
            s += m;
        }
    vec[idx] = ((s * (1.f / 49.f)) >= 0.f) ? FP8_P1 : FP8_M1;
}

__global__ void packdense_kernel(const float* __restrict__ dw, uint8_t* __restrict__ out)
{
    int k = blockIdx.x * 256 + threadIdx.x;
    int n = blockIdx.y;
    if (k >= 512) return;
    uint8_t byte = 0;
    if (n < 1000) byte = (dw[(size_t)k * 1000 + n] >= 0.f) ? FP8_P1 : FP8_M1;
    out[(size_t)n * 512 + k] = byte;
}

__global__ __launch_bounds__(32)
void bdense_kernel(const uint8_t* __restrict__ a8, const uint8_t* __restrict__ w8,
                   float* __restrict__ out)
{
    int lane = threadIdx.x & 31;
    int hf = lane >> 4, l15 = lane & 15;
    int n0 = blockIdx.x * 16;

    v8f acc = {0.f, 0.f, 0.f, 0.f, 0.f, 0.f, 0.f, 0.f};
    for (int ch = 0; ch < 4; ++ch) {
        v16i a, bf;
        const uint8_t* ap = a8 + (size_t)l15 * 512 + ch * 128 + 16 * hf;
        #pragma unroll
        for (int j = 0; j < 8; ++j) {
            uint2 q = *(const uint2*)(ap + 32 * (j >> 1) + 8 * (j & 1));
            a[2 * j] = (int)q.x; a[2 * j + 1] = (int)q.y;
        }
        const uint8_t* bp = w8 + (size_t)(n0 + l15) * 512 + ch * 128 + 16 * hf;
        #pragma unroll
        for (int g = 0; g < 4; ++g) {
            uint4 q = *(const uint4*)(bp + 32 * g);
            bf[4 * g] = (int)q.x; bf[4 * g + 1] = (int)q.y;
            bf[4 * g + 2] = (int)q.z; bf[4 * g + 3] = (int)q.w;
        }
        acc = __builtin_amdgcn_wmma_f32_16x16x128_fp8_fp8(a, bf, (short)0, acc,
                                                          false, false);
    }
    #pragma unroll
    for (int r = 0; r < 8; ++r) {
        int m = 8 * hf + r;
        int n = n0 + l15;
        if (n < 1000) out[(size_t)m * 1000 + n] = acc[r];
    }
}

extern "C" void kernel_launch(void* const* d_in, const int* in_sizes, int n_in,
                              void* d_out, int out_size, void* d_ws, size_t ws_size,
                              hipStream_t stream)
{
    (void)in_sizes; (void)n_in; (void)out_size; (void)ws_size;

    const float* inputs  = (const float*)d_in[0];
    const float* conv1w  = (const float*)d_in[1];
    const float* conv1b  = (const float*)d_in[2];
    const float* bn1     = (const float*)d_in[3];
    const float* wA[9]   = {nullptr, nullptr,
                            (const float*)d_in[4],  (const float*)d_in[6],
                            (const float*)d_in[8],  (const float*)d_in[10],
                            (const float*)d_in[12], (const float*)d_in[14],
                            (const float*)d_in[16]};
    const float* bnA[9]  = {nullptr, nullptr,
                            (const float*)d_in[5],  (const float*)d_in[7],
                            (const float*)d_in[9],  (const float*)d_in[11],
                            (const float*)d_in[13], (const float*)d_in[15],
                            (const float*)d_in[17]};
    const float* densew  = (const float*)d_in[18];
    float*       out     = (float*)d_out;

    uint8_t* ws  = (uint8_t*)d_ws;
    uint8_t* P0  = ws;
    uint8_t* P1  = ws + (13u << 20);
    uint8_t* WP  = ws + (40u << 20);
    float*   Y8  = (float*)(ws + (45u << 20));
    uint8_t* VEC = ws + (54u << 20);
    uint8_t* DW  = ws + (54u << 20) + 8192;

    const int B = 16;

    auto bconv = [&](const uint8_t* inp, int layer, uint8_t* o8, float* of,
                     int H, int W, int cinLog2, int Cout, bool fp32out) {
        int Cin  = 1 << cinLog2;
        int Kpad = ((9 * Cin + 127) / 128) * 128;
        dim3 pg((Kpad + 255) / 256, Cout);
        packw_kernel<<<pg, 256, 0, stream>>>(wA[layer], WP, Cin, cinLog2, Cout, Kpad);
        int M = B * H * W;
        dim3 g((M + 127) / 128, Cout / 64);
        if (fp32out)
            bconv_kernel<true><<<g, 256, 0, stream>>>(inp, WP, bnA[layer], nullptr, of,
                                                      H, W, cinLog2, Cout, Kpad, M);
        else
            bconv_kernel<false><<<g, 256, 0, stream>>>(inp, WP, bnA[layer], o8, nullptr,
                                                       H, W, cinLog2, Cout, Kpad, M);
    };
    auto pool = [&](const uint8_t* inp, uint8_t* o, int Ho, int Wo, int C) {
        size_t total = (size_t)B * Ho * Wo * C;
        pool8_kernel<<<(unsigned)((total + 255) / 256), 256, 0, stream>>>(inp, o, Ho, Wo, C, total);
    };

    stage1_kernel<<<B * 112 * 112, 64, 0, stream>>>(inputs, conv1w, conv1b, bn1, P0);

    bconv(P0, 2, P1, nullptr, 112, 112, 6, 128, false);
    pool (P1, P0, 56, 56, 128);
    bconv(P0, 3, P1, nullptr, 56, 56, 7, 256, false);
    bconv(P1, 4, P0, nullptr, 56, 56, 8, 256, false);
    pool (P0, P1, 28, 28, 256);
    bconv(P1, 5, P0, nullptr, 28, 28, 8, 512, false);
    bconv(P0, 6, P1, nullptr, 28, 28, 9, 512, false);
    pool (P1, P0, 14, 14, 512);
    bconv(P0, 7, P1, nullptr, 14, 14, 9, 512, false);
    bconv(P1, 8, nullptr, Y8, 14, 14, 9, 512, true);

    gap_kernel<<<32, 256, 0, stream>>>(Y8, VEC);
    packdense_kernel<<<dim3(2, 1008), 256, 0, stream>>>(densew, DW);
    bdense_kernel<<<63, 32, 0, stream>>>(VEC, DW, out);
}
